// ArcFaceLoss_66864050864571
// MI455X (gfx1250) — matrix operand order repaired and checked by reading, not compiled, not run on hardware
//
#include <hip/hip_runtime.h>
#include <hip/hip_bf16.h>
#include <math.h>

// ---------------------------------------------------------------------------
// ArcFace loss for MI455X (gfx1250, wave32, WMMA).
// cosine GEMM done as split-precision bf16 (hi+lo) with v_wmma_f32_16x16x32_bf16:
//   a.b ~= aHi.bHi + aHi.bLo + aLo.bHi   (fp32-class accuracy, 8x K-depth vs f32 WMMA)
// GEMM body is branch-free (clamped edge tiles) so EXEC stays all-ones around
// every WMMA; all loads/stores use 32-bit offsets off SGPR bases with the fully
// unrolled K folded into load immediates.
// ---------------------------------------------------------------------------

typedef __attribute__((ext_vector_type(16))) __bf16     v16bf;
typedef __attribute__((ext_vector_type(8)))  float      v8f;
typedef __attribute__((ext_vector_type(4)))  unsigned   u32x4;

union BFrag { v16bf v; u32x4 q[2]; };

static constexpr int   kD      = 512;
static constexpr float kScale  = 64.0f;
static constexpr float kCosM   = 0.87758256189037276f;   // cos(0.5)
static constexpr float kSinM   = 0.47942553860420301f;   // sin(0.5)
static constexpr float kTH     = -0.87758256189037276f;  // cos(pi-0.5)
static constexpr float kMM     = 0.23971276930210151f;   // sin(pi-0.5)*0.5

// ---------------------------------------------------------------------------
// Kernel 1: per-row L2 normalize + split fp32 -> (bf16 hi, bf16 lo).
// One wave (32 lanes) per row of D=512.
// ---------------------------------------------------------------------------
__global__ __launch_bounds__(256)
void normalize_split_kernel(const float* __restrict__ src,
                            __bf16* __restrict__ hiOut,
                            __bf16* __restrict__ loOut,
                            int rows) {
    int wave = (blockIdx.x * blockDim.x + threadIdx.x) >> 5;
    int lane = threadIdx.x & 31;
    if (wave >= rows) return;

    const float* p = src + (size_t)wave * kD;
    float ss = 0.0f;
#pragma unroll
    for (int i = lane; i < kD; i += 32) {
        float v = p[i];
        ss += v * v;
    }
#pragma unroll
    for (int off = 16; off > 0; off >>= 1)
        ss += __shfl_xor(ss, off, 32);

    float inv = 1.0f / fmaxf(sqrtf(ss), 1e-12f);

    __bf16* ho = hiOut + (size_t)wave * kD;
    __bf16* lo = loOut + (size_t)wave * kD;
#pragma unroll
    for (int i = lane; i < kD; i += 32) {
        float v  = p[i] * inv;
        __bf16 h = (__bf16)v;            // RNE
        ho[i] = h;
        lo[i] = (__bf16)(v - (float)h);  // residual
    }
}

// ---------------------------------------------------------------------------
// Kernel 2: split-bf16 WMMA GEMM.  out[b][c] = 64 * <embN[b], wN[c]>.
// Block = 256 threads = 8 waves arranged 4(M) x 2(N).
// Wave tile = 16M x 64N (4 accumulators, A-fragment reused 4x).
// Block tile = 64M x 128N.  K = 512 in steps of 32.
// Requires: M % 64 == 0, N % 16 == 0 (true for 1024 x 50000).
//
// Fragment layouts follow CDNA5 ISA 7.12.2 (wave32):
//   A 16x32 bf16: lane m(+16 for K-half h): VGPR0-3 = K[h*8 .. h*8+7],
//                 VGPR4-7 = K[16+h*8 .. 16+h*8+7]  (two contiguous 16B runs)
//   B 32x16 bf16: lane n(+16 for K-half h): K[h*8 .. h*8+7], K[16+h*8 .. 16+h*8+7] of column n
//                 (two contiguous 16B runs of the weight row)
//   C/D f32: VGPR r holds M = r + 8*(lane/16), N = lane%16
//
// hi/lo arrays share one layout, so ONE 32-bit element offset addresses both;
// output stores likewise use one 32-bit offset per tile (r*N strength-reduced).
// All 32-bit offsets fit: max 51.2M elements / 204.8MB bytes < 2^31.
// ---------------------------------------------------------------------------
__global__ __launch_bounds__(256)
void arcface_gemm_kernel(const __bf16* __restrict__ aHi,
                         const __bf16* __restrict__ aLo,
                         const __bf16* __restrict__ bHi,
                         const __bf16* __restrict__ bLo,
                         float* __restrict__ outC,
                         int N) {
    const int w    = threadIdx.x >> 5;
    const int lane = threadIdx.x & 31;
    const int mw   = w & 3;          // 0..3  M sub-block
    const int nw   = w >> 2;         // 0..1  N sub-block
    const int rowBase = blockIdx.y * 64 + mw * 16;
    const int colBase = blockIdx.x * 128 + nw * 64;

    const int m = lane & 15;         // row within A tile / col within B tile
    const int h = lane >> 4;         // K-half selector

    // Clamp each 16-col subtile into range (branch-free ragged edge: the
    // clamped tiles recompute identical values and overwrite identically).
    int c0[4];
#pragma unroll
    for (int t = 0; t < 4; ++t) {
        int c = colBase + t * 16;
        c0[t] = (c > N - 16) ? (N - 16) : c;
    }

    // 32-bit element offsets (shared by the hi and lo arrays).
    const unsigned aOff = ((unsigned)(rowBase + m) << 9) + (unsigned)(h * 8);
    unsigned bOff[4];
#pragma unroll
    for (int t = 0; t < 4; ++t)
        bOff[t] = ((unsigned)(c0[t] + m) << 9) + (unsigned)(h * 8);

    v8f acc[4] = {};

#pragma unroll
    for (int kk = 0; kk < kD; kk += 32) {
        BFrag ah, al;
        ah.q[0] = *reinterpret_cast<const u32x4*>(aHi + aOff + kk);
        ah.q[1] = *reinterpret_cast<const u32x4*>(aHi + aOff + kk + 16);
        al.q[0] = *reinterpret_cast<const u32x4*>(aLo + aOff + kk);
        al.q[1] = *reinterpret_cast<const u32x4*>(aLo + aOff + kk + 16);

        BFrag bh[4], bl[4];
#pragma unroll
        for (int t = 0; t < 4; ++t) {
            bh[t].q[0] = *reinterpret_cast<const u32x4*>(bHi + bOff[t] + kk);
            bh[t].q[1] = *reinterpret_cast<const u32x4*>(bHi + bOff[t] + kk + 16);
            bl[t].q[0] = *reinterpret_cast<const u32x4*>(bLo + bOff[t] + kk);
            bl[t].q[1] = *reinterpret_cast<const u32x4*>(bLo + bOff[t] + kk + 16);
        }

        // Light prefetch: every 4th K-step, pull the next 256B of each B stream.
        if ((kk & 127) == 0 && kk + 128 < kD) {
#pragma unroll
            for (int t = 0; t < 4; ++t) {
                __builtin_prefetch(bHi + bOff[t] + kk + 128, 0, 3);  // global_prefetch_b8
                __builtin_prefetch(bLo + bOff[t] + kk + 128, 0, 3);
            }
        }

        // Three passes over 4 independent accumulators: consecutive WMMAs never
        // chain on the same C operand -> XDL pipe stays busy.
#pragma unroll
        for (int t = 0; t < 4; ++t)
            acc[t] = __builtin_amdgcn_wmma_f32_16x16x32_bf16(
                false, ah.v, false, bh[t].v, (short)0, acc[t], false, false);
#pragma unroll
        for (int t = 0; t < 4; ++t)
            acc[t] = __builtin_amdgcn_wmma_f32_16x16x32_bf16(
                false, ah.v, false, bl[t].v, (short)0, acc[t], false, false);
#pragma unroll
        for (int t = 0; t < 4; ++t)
            acc[t] = __builtin_amdgcn_wmma_f32_16x16x32_bf16(
                false, al.v, false, bh[t].v, (short)0, acc[t], false, false);
    }

    // Epilogue: one 32-bit element offset per tile, r*N strength-reduced.
    const unsigned uN = (unsigned)N;
#pragma unroll
    for (int t = 0; t < 4; ++t) {
        unsigned oOff = (unsigned)(rowBase + h * 8) * uN + (unsigned)(c0[t] + m);
#pragma unroll
        for (int r = 0; r < 8; ++r) {
            outC[oOff] = acc[t][r] * kScale;
            oOff += uN;
        }
    }
}

// ---------------------------------------------------------------------------
// Kernel 3: zero the loss accumulator.
// ---------------------------------------------------------------------------
__global__ void init_loss_kernel(float* lossOut) { *lossOut = 0.0f; }

// ---------------------------------------------------------------------------
// Kernel 4: per-row online logsumexp over the (phi-substituted) logits,
// accumulate mean(logz - true_logit) into lossOut.
// One 256-thread block per batch row.
// ---------------------------------------------------------------------------
__global__ __launch_bounds__(256)
void arcface_loss_kernel(const float* __restrict__ cosScaled,  // [B,N], already *64
                         const int* __restrict__ labels,
                         float* __restrict__ lossOut,
                         int N, int B) {
    const int b   = blockIdx.x;
    const int tid = threadIdx.x;
    const int lbl = labels[b];
    const unsigned rowBase = (unsigned)b * (unsigned)N;   // < 2^31 elements

    float mmax = -INFINITY;
    float msum = 0.0f;
    for (int c = tid; c < N; c += 256) {
        float v = cosScaled[rowBase + (unsigned)c];
        if (c == lbl) {
            float cs  = v * (1.0f / kScale);
            float sn  = sqrtf(fmaxf(1.0f - cs * cs, 0.0f));
            float phi = cs * kCosM - sn * kSinM;
            if (!(cs > kTH)) phi = cs - kMM;
            v = phi * kScale;
        }
        if (v > mmax) {
            msum = msum * __expf(mmax - v) + 1.0f;
            mmax = v;
        } else {
            msum += __expf(v - mmax);
        }
    }

    __shared__ float smax[256];
    __shared__ float ssum[256];
    smax[tid] = mmax;
    ssum[tid] = msum;
    __syncthreads();
#pragma unroll
    for (int off = 128; off > 0; off >>= 1) {
        if (tid < off) {
            float m1 = smax[tid],       s1 = ssum[tid];
            float m2 = smax[tid + off], s2 = ssum[tid + off];
            float mh = fmaxf(m1, m2);
            float sh = 0.0f;
            if (m1 > -INFINITY) sh += s1 * __expf(m1 - mh);
            if (m2 > -INFINITY) sh += s2 * __expf(m2 - mh);
            smax[tid] = mh;
            ssum[tid] = sh;
        }
        __syncthreads();
    }

    if (tid == 0) {
        float logz = smax[0] + logf(ssum[0]);
        float cs   = cosScaled[rowBase + (unsigned)lbl] * (1.0f / kScale);
        float sn   = sqrtf(fmaxf(1.0f - cs * cs, 0.0f));
        float phi  = cs * kCosM - sn * kSinM;
        if (!(cs > kTH)) phi = cs - kMM;
        float trueLogit = phi * kScale;
        atomicAdd(lossOut, (logz - trueLogit) / (float)B);
    }
}

// ---------------------------------------------------------------------------
// Launcher. d_in: [embeddings fp32 BxD, labels i32 B, weight fp32 CxD]
// d_out: [loss (1 fp32)] ++ [cosine*64 (BxC fp32)]
// Workspace: emb hi/lo bf16 (2 * B*D) then weight hi/lo bf16 (2 * C*D).
// ---------------------------------------------------------------------------
extern "C" void kernel_launch(void* const* d_in, const int* in_sizes, int n_in,
                              void* d_out, int out_size, void* d_ws, size_t ws_size,
                              hipStream_t stream) {
    const float* emb    = (const float*)d_in[0];
    const int*   labels = (const int*)d_in[1];
    const float* weight = (const float*)d_in[2];

    const int B = in_sizes[1];          // 1024
    const int C = in_sizes[2] / kD;     // 50000

    float* out = (float*)d_out;

    __bf16* embHi = (__bf16*)d_ws;
    __bf16* embLo = embHi + (size_t)B * kD;
    __bf16* wHi   = embLo + (size_t)B * kD;
    __bf16* wLo   = wHi   + (size_t)C * kD;

    // 1) normalize + split (one wave per row, 8 waves per block)
    normalize_split_kernel<<<(B + 7) / 8, 256, 0, stream>>>(emb, embHi, embLo, B);
    normalize_split_kernel<<<(C + 7) / 8, 256, 0, stream>>>(weight, wHi, wLo, C);

    // 2) loss accumulator init
    init_loss_kernel<<<1, 1, 0, stream>>>(out);

    // 3) WMMA GEMM -> cosine*64 into out+1
    dim3 grid((C + 127) / 128, (B + 63) / 64);
    arcface_gemm_kernel<<<grid, 256, 0, stream>>>(embHi, embLo, wHi, wLo,
                                                  out + 1, C);

    // 4) logsumexp + loss
    arcface_loss_kernel<<<B, 256, 0, stream>>>(out + 1, labels, out, C, B);
}
